// Attention_52604759441789
// MI455X (gfx1250) — matrix operand order repaired and checked by reading, not compiled, not run on hardware
//
#include <hip/hip_runtime.h>
#include <cstdint>

// ---------------------------------------------------------------------------
// MI455X (gfx1250) attention layer: bf16 WMMA GEMMs + RoPE + flash attention.
// Memory-bound (~0.4 GB traffic, ~0.69 TFLOP) -> bf16 everywhere, all matmuls
// on v_wmma_f32_16x16x32_bf16 (wave32), global->LDS via CDNA5 async copies
// (ASYNCcnt) with double-buffered LDS so the HBM stream overlaps the WMMAs.
// ---------------------------------------------------------------------------

typedef __bf16 bf16_t;
typedef __attribute__((ext_vector_type(16))) __bf16 v16bf;
typedef __attribute__((ext_vector_type(8)))  float  v8f;
typedef __attribute__((__vector_size__(16))) int    v4i_raw;  // builtin pointee

#ifndef __has_builtin
#define __has_builtin(x) 0
#endif
#if __has_builtin(__builtin_amdgcn_global_load_async_to_lds_b128)
#define USE_ASYNC_LDS 1
#else
#define USE_ASYNC_LDS 0
#endif

union ABFrag {
  v16bf  v;
  uint4  q[2];
  bf16_t b[16];
};

__device__ __forceinline__ v8f zero_v8f() {
  v8f z;
#pragma unroll
  for (int i = 0; i < 8; ++i) z[i] = 0.0f;
  return z;
}

__device__ __forceinline__ v8f wmma_bf16(v16bf a, v16bf b, v8f c) {
  // (neg_a, A, neg_b, B, c_mod, C, reuse_a, reuse_b)
  return __builtin_amdgcn_wmma_f32_16x16x32_bf16(false, a, false, b,
                                                 (short)0, c, false, false);
}

// copy 32 B (16 bf16) global -> LDS; async path uses ASYNCcnt-tracked DMA.
// Builtin signature (from hipcc diagnostic): (v4i* global, v4i* lds, Ioff, cpol)
// with generic pointers implicitly converted to the __device__/__shared__ AS.
__device__ __forceinline__ void copy32B(bf16_t* l, const bf16_t* g) {
#if USE_ASYNC_LDS
  __builtin_amdgcn_global_load_async_to_lds_b128((v4i_raw*)g, (v4i_raw*)l, 0, 0);
  __builtin_amdgcn_global_load_async_to_lds_b128((v4i_raw*)(g + 8),
                                                 (v4i_raw*)(l + 8), 0, 0);
#else
  uint4 t0 = *(const uint4*)g;
  uint4 t1 = *(const uint4*)(g + 8);
  *(uint4*)l = t0;
  *(uint4*)(l + 8) = t1;
#endif
}

__device__ __forceinline__ void async_wait0() {
#if USE_ASYNC_LDS
#if __has_builtin(__builtin_amdgcn_s_wait_asynccnt)
  __builtin_amdgcn_s_wait_asynccnt(0);
#else
  asm volatile("s_wait_asynccnt 0x0" ::: "memory");
#endif
#endif
}

// ---------------- fp32 -> bf16 stream conversion (4 elems/thread) ----------
__global__ void cast_f32_to_bf16(const float* __restrict__ in,
                                 bf16_t* __restrict__ out, int n4) {
  int i = blockIdx.x * blockDim.x + threadIdx.x;
  int stride = gridDim.x * blockDim.x;
  for (; i < n4; i += stride) {
    float4 f = ((const float4*)in)[i];
    union { bf16_t b[4]; uint2 u; } p;
    p.b[0] = (bf16_t)f.x; p.b[1] = (bf16_t)f.y;
    p.b[2] = (bf16_t)f.z; p.b[3] = (bf16_t)f.w;
    ((uint2*)out)[i] = p.u;
  }
}

// ---------------- GEMM: C[M,N] = A[M,K] * W[N,K]^T  (bf16 in, OutT out) ----
// Block tile 128x128, K-step 32, double-buffered LDS fed by async copies.
// 8 waves (2 Mw x 4 Nw), each wave 64x32 via 4x2 WMMA accumulators.
template <typename OutT>
__global__ __launch_bounds__(256)
void gemm_bf16_wmma(const bf16_t* __restrict__ A,
                    const bf16_t* __restrict__ W,
                    OutT* __restrict__ C,
                    int M, int N, int K) {
  constexpr int LDT = 40;                 // 32 + 8 pad (bf16) per LDS row
  __shared__ bf16_t lA[2][128 * LDT];
  __shared__ bf16_t lB[2][128 * LDT];
  const int tid  = threadIdx.x;
  const int lane = tid & 31, wave = tid >> 5;
  const int hl   = lane >> 4, l16 = lane & 15;   // half-wave, lane-in-16
  const int wm   = wave & 1,  wn  = wave >> 1;
  const int m0   = blockIdx.y * 128, n0 = blockIdx.x * 128;

  v8f acc[4][2];
#pragma unroll
  for (int i = 0; i < 4; ++i)
#pragma unroll
    for (int j = 0; j < 2; ++j) acc[i][j] = zero_v8f();

  const int ldr = tid >> 1;              // 0..127 tile row
  const int ldc = (tid & 1) * 16;        // which 32B half of the 64B row
  const bf16_t* ga = A + (long)(m0 + ldr) * K + ldc;
  const bf16_t* gb = W + (long)(n0 + ldr) * K + ldc;
  const int lo = ldr * LDT + ldc;

  // prologue: stage K-tile 0 into buffer 0
  copy32B(&lA[0][lo], ga);
  copy32B(&lB[0][lo], gb);
  async_wait0();
  __syncthreads();

  const int nk = K >> 5;
  int buf = 0;
  for (int kc = 0; kc < nk; ++kc) {
    if (kc + 1 < nk) {  // stage next K-tile into the other buffer (async)
      copy32B(&lA[buf ^ 1][lo], ga + (kc + 1) * 32);
      copy32B(&lB[buf ^ 1][lo], gb + (kc + 1) * 32);
    }

    ABFrag fa[4], fb[2];
#pragma unroll
    for (int i = 0; i < 4; ++i) {
      // A layout (ISA 7.12.2): lane M = l16, elems K = (j/8)*16 + hl*8 + j%8
      const bf16_t* p = &lA[buf][(wm * 64 + i * 16 + l16) * LDT + hl * 8];
      fa[i].q[0] = *(const uint4*)p;
      fa[i].q[1] = *(const uint4*)(p + 16);
    }
#pragma unroll
    for (int j = 0; j < 2; ++j) {
      // B layout: lane N = l16, elems K = (j/8)*16 + hl*8 + j%8
      const bf16_t* p = &lB[buf][(wn * 32 + j * 16 + l16) * LDT + hl * 8];
      fb[j].q[0] = *(const uint4*)p;
      fb[j].q[1] = *(const uint4*)(p + 16);
    }
#pragma unroll
    for (int i = 0; i < 4; ++i)
#pragma unroll
      for (int j = 0; j < 2; ++j)
        acc[i][j] = wmma_bf16(fa[i].v, fb[j].v, acc[i][j]);

    async_wait0();      // next tile fully in LDS
    __syncthreads();    // everyone done reading current buffer
    buf ^= 1;
  }

  // C layout: lane N = l16, VGPR r -> M = r + 8*hl
#pragma unroll
  for (int i = 0; i < 4; ++i)
#pragma unroll
    for (int j = 0; j < 2; ++j) {
      long row = m0 + wm * 64 + i * 16;
      long col = n0 + wn * 32 + j * 16 + l16;
#pragma unroll
      for (int r = 0; r < 8; ++r)
        C[(row + r + 8 * hl) * (long)N + col] = (OutT)acc[i][j][r];
    }
}

// ---------------- interleaved RoPE on the packed (b,s,2,h,d) QK buffer -----
__global__ void rope_bf16(bf16_t* __restrict__ qk) {
  int i = blockIdx.x * blockDim.x + threadIdx.x;  // 2^24 pairs
  int p   = i & 63;
  int h   = (i >> 6) & 31;
  int sel = (i >> 11) & 1;
  int s   = (i >> 12) & 2047;
  int b   = (i >> 23) & 1;
  long base = ((long)(b * 2048 + s)) * 8192 + sel * 4096 + h * 128 + 2 * p;
  float inv = __powf(10000.0f, -(float)(2 * p) * (1.0f / 128.0f));
  float ang = (float)s * inv;
  float sn, cs;
  __sincosf(ang, &sn, &cs);
  float e = (float)qk[base], o = (float)qk[base + 1];
  qk[base]     = (bf16_t)(e * cs - o * sn);
  qk[base + 1] = (bf16_t)(e * sn + o * cs);
}

// ---------------- flash attention: causal, online softmax, all-WMMA --------
// grid = (S/128, B*H). 8 waves/block; wave w owns query rows q0+16w..+15 with
// Q resident in registers. Keys in chunks of 32: K staged row-major (async
// copies), V staged transposed, next chunk prefetched (global_prefetch_b8).
__global__ __launch_bounds__(256)
void flash_attn_bf16(const bf16_t* __restrict__ qk,
                     const bf16_t* __restrict__ vbuf,
                     bf16_t* __restrict__ obuf) {
  constexpr int S = 2048;
  constexpr int KLD = 136;  // 128 + 8 pad (bf16)
  constexpr int VLD = 40;   // 32 + 8 pad  (bf16)
  __shared__ bf16_t lK[32 * KLD];    // [key][d]
  __shared__ bf16_t lVt[128 * VLD];  // [d][key]
  __shared__ bf16_t lP[128 * VLD];   // per-wave 16-row P staging

  const int tid = threadIdx.x, lane = tid & 31, wave = tid >> 5;
  const int hl = lane >> 4, l16 = lane & 15;
  const int b = blockIdx.y >> 5, h = blockIdx.y & 31;
  const int q0 = blockIdx.x * 128;

  // preload Q fragments: 16 rows x 128 d -> 4 A-fragments
  ABFrag qf[4];
  {
    const bf16_t* qrow =
        qk + ((long)(b * S + q0 + wave * 16 + l16)) * 8192 + h * 128;
#pragma unroll
    for (int c = 0; c < 4; ++c) {
      qf[c].q[0] = *(const uint4*)(qrow + c * 32 + hl * 8);
      qf[c].q[1] = *(const uint4*)(qrow + c * 32 + 16 + hl * 8);
    }
  }

  v8f ov[8];
  float mrow[8], lrow[8];
#pragma unroll
  for (int r = 0; r < 8; ++r) {
    ov[r] = zero_v8f();
    mrow[r] = -3.0e38f;
    lrow[r] = 0.0f;
  }

  const int nchunk = q0 / 32 + 4;            // keys up to q0+127 inclusive
  const float scale = 0.08838834764831845f;  // 1/sqrt(128)

  for (int kc = 0; kc < nchunk; ++kc) {
    const int kp0 = kc * 32;
    {  // cooperative staging: 32B of K (async) + 32B of V (transposed)
      int r  = tid >> 3;
      int d0 = (tid & 7) * 16;
      const bf16_t* gk =
          qk + ((long)(b * S + kp0 + r)) * 8192 + 4096 + h * 128 + d0;
      copy32B(&lK[r * KLD + d0], gk);
      const bf16_t* gv =
          vbuf + ((long)(b * S + kp0 + r)) * 4096 + h * 128 + d0;
      ABFrag t;
      t.q[0] = *(const uint4*)gv;
      t.q[1] = *(const uint4*)(gv + 8);
#pragma unroll
      for (int e = 0; e < 16; ++e) lVt[(d0 + e) * VLD + r] = t.b[e];
      if (kc + 1 < nchunk) {  // prefetch next chunk (global_prefetch_b8)
        __builtin_prefetch(gk + 32 * 8192, 0, 0);
        __builtin_prefetch(gv + 32 * 4096, 0, 0);
      }
    }
    async_wait0();
    __syncthreads();

    if (kp0 <= q0 + wave * 16 + 15) {  // wave-uniform causal skip
      // S tile: 16 q-rows x 32 keys = two 16x16 WMMA outputs
      v8f s0 = zero_v8f(), s1 = zero_v8f();
#pragma unroll
      for (int c = 0; c < 4; ++c) {
        ABFrag kf;
        const bf16_t* kptr0 = &lK[l16 * KLD + c * 32 + hl * 8];
        kf.q[0] = *(const uint4*)kptr0;
        kf.q[1] = *(const uint4*)(kptr0 + 16);
        s0 = wmma_bf16(qf[c].v, kf.v, s0);
        const bf16_t* kptr1 = &lK[(16 + l16) * KLD + c * 32 + hl * 8];
        kf.q[0] = *(const uint4*)kptr1;
        kf.q[1] = *(const uint4*)(kptr1 + 16);
        s1 = wmma_bf16(qf[c].v, kf.v, s1);
      }
      // online softmax: row M = r + 8*hl, column N = l16 (C layout)
#pragma unroll
      for (int r = 0; r < 8; ++r) {
        int qpos = q0 + wave * 16 + r + 8 * hl;
        float v0 = (kp0 + l16      <= qpos) ? s0[r] * scale : -3.0e38f;
        float v1 = (kp0 + 16 + l16 <= qpos) ? s1[r] * scale : -3.0e38f;
        float mx = fmaxf(v0, v1);
#pragma unroll
        for (int off = 8; off; off >>= 1)
          mx = fmaxf(mx, __shfl_xor(mx, off, 16));
        float mnew = fmaxf(mrow[r], mx);
        float corr = __expf(mrow[r] - mnew);
        float e0 = __expf(v0 - mnew);
        float e1 = __expf(v1 - mnew);
        float ps = e0 + e1;
#pragma unroll
        for (int off = 8; off; off >>= 1) ps += __shfl_xor(ps, off, 16);
        lrow[r] = lrow[r] * corr + ps;
        mrow[r] = mnew;
#pragma unroll
        for (int t = 0; t < 8; ++t) ov[t][r] *= corr;
        int prow = wave * 16 + r + 8 * hl;  // per-wave private LDS rows
        lP[prow * VLD + l16]      = (bf16_t)e0;
        lP[prow * VLD + 16 + l16] = (bf16_t)e1;
      }
      // O += P * V : P re-read in A layout (same-wave LDS ops are in-order)
      ABFrag pf;
      const bf16_t* pp = &lP[(wave * 16 + l16) * VLD + hl * 8];
      pf.q[0] = *(const uint4*)pp;
      pf.q[1] = *(const uint4*)(pp + 16);
#pragma unroll
      for (int t = 0; t < 8; ++t) {
        ABFrag vf;
        const bf16_t* vp = &lVt[(t * 16 + l16) * VLD + hl * 8];
        vf.q[0] = *(const uint4*)vp;
        vf.q[1] = *(const uint4*)(vp + 16);
        ov[t] = wmma_bf16(pf.v, vf.v, ov[t]);
      }
    }
    __syncthreads();
  }

#pragma unroll
  for (int r = 0; r < 8; ++r) {
    float invl = 1.0f / lrow[r];
    bf16_t* op =
        obuf + ((long)(b * S + q0 + wave * 16 + r + 8 * hl)) * 4096 + h * 128;
#pragma unroll
    for (int t = 0; t < 8; ++t)
      op[t * 16 + l16] = (bf16_t)(ov[t][r] * invl);
  }
}

// ---------------------------------------------------------------------------
extern "C" void kernel_launch(void* const* d_in, const int* in_sizes, int n_in,
                              void* d_out, int out_size, void* d_ws,
                              size_t ws_size, hipStream_t stream) {
  const float* x   = (const float*)d_in[0];
  const float* wqk = (const float*)d_in[1];
  const float* wv  = (const float*)d_in[2];
  const float* wo  = (const float*)d_in[3];
  float* out = (float*)d_out;

  const int  Bt = 2, S = 2048, D = 4096;
  const long MT = (long)Bt * S;  // 4096 token rows

  char* ws = (char*)d_ws;
  size_t off = 0;
  auto carve = [&](size_t bytes) -> void* {
    void* p = ws + off;
    off = (off + bytes + 255) & ~(size_t)255;
    return p;
  };
  // ~302 MB workspace: bf16 copies of x/weights + qk/v/attn-out activations
  bf16_t* xb   = (bf16_t*)carve((size_t)MT * D * 2);
  bf16_t* wqkb = (bf16_t*)carve((size_t)2 * D * D * 2);
  bf16_t* wvb  = (bf16_t*)carve((size_t)D * D * 2);
  bf16_t* wob  = (bf16_t*)carve((size_t)D * D * 2);
  bf16_t* qkb  = (bf16_t*)carve((size_t)MT * 2 * D * 2);
  bf16_t* vb   = (bf16_t*)carve((size_t)MT * D * 2);
  bf16_t* aob  = (bf16_t*)carve((size_t)MT * D * 2);
  (void)ws_size; (void)in_sizes; (void)n_in; (void)out_size;

  auto cast = [&](const float* src, bf16_t* dst, long n) {
    int n4 = (int)(n / 4);
    int grid = (n4 + 255) / 256;
    if (grid > 65535) grid = 65535;
    cast_f32_to_bf16<<<dim3(grid), dim3(256), 0, stream>>>(src, dst, n4);
  };
  cast(x, xb, MT * D);
  cast(wqk, wqkb, 2L * D * D);
  cast(wv, wvb, (long)D * D);
  cast(wo, wob, (long)D * D);

  // QK projection: [4096 x 8192] = xb * wqkb^T
  gemm_bf16_wmma<bf16_t><<<dim3(8192 / 128, 4096 / 128), dim3(256), 0, stream>>>(
      xb, wqkb, qkb, 4096, 8192, 4096);
  // V projection: [4096 x 4096]
  gemm_bf16_wmma<bf16_t><<<dim3(4096 / 128, 4096 / 128), dim3(256), 0, stream>>>(
      xb, wvb, vb, 4096, 4096, 4096);
  // RoPE (in place on qkb)
  rope_bf16<<<dim3((1 << 24) / 256), dim3(256), 0, stream>>>(qkb);
  // causal flash attention
  flash_attn_bf16<<<dim3(S / 128, Bt * 32), dim3(256), 0, stream>>>(qkb, vb, aob);
  // output projection straight to fp32 d_out
  gemm_bf16_wmma<float><<<dim3(4096 / 128, 4096 / 128), dim3(256), 0, stream>>>(
      aob, wob, out, 4096, 4096, 4096);
}
